// CausalSelfAttention_15479062134845
// MI455X (gfx1250) — matrix operand order repaired and checked by reading, not compiled, not run on hardware
//
#include <hip/hip_runtime.h>

// ---------------------------------------------------------------------------
// CausalSelfAttention forward on gfx1250 (MI455X, wave32, WMMA + TDM)
// B=2, T=2048, C=1024, H=16, D=64
// Pipeline: f32->bf16 convert, QKV GEMM (TDM-staged LDS tiles, bf16 WMMA),
//           flash attention (S^T / O^T formulation, bf16 WMMA, f32 acc)
// ---------------------------------------------------------------------------

#define BATCH 2
#define SEQ   2048
#define CDIM  1024
#define HEADS 16
#define HDIM  64
#define NROWS (BATCH * SEQ)     // 4096
#define FDIM  (3 * CDIM)        // 3072

typedef __attribute__((ext_vector_type(16))) __bf16        v16bf;
typedef __attribute__((ext_vector_type(8)))  float         v8f;
typedef __attribute__((ext_vector_type(8)))  unsigned int  v8u;
typedef __attribute__((ext_vector_type(4)))  unsigned int  v4u;
typedef __attribute__((ext_vector_type(8)))  int           v8i;
typedef __attribute__((ext_vector_type(4)))  int           v4i;

__device__ __forceinline__ unsigned short f2bf_u(float f) {
  unsigned u = __builtin_bit_cast(unsigned, f);
  unsigned r = u + 0x7FFFu + ((u >> 16) & 1u);   // round-to-nearest-even
  return (unsigned short)(r >> 16);
}

__device__ __forceinline__ v16bf pack_frag(uint4 a, uint4 b) {
  v8u t;
  t[0] = a.x; t[1] = a.y; t[2] = a.z; t[3] = a.w;
  t[4] = b.x; t[5] = b.y; t[6] = b.z; t[7] = b.w;
  return __builtin_bit_cast(v16bf, t);
}

// ---------------------------------------------------------------------------
// Tensor Data Mover: async 2D tile load global -> LDS.
// Builds the D# descriptor per CDNA5 ISA (CDNA5 ISA):
//  group0: [1:0]=count=1, [63:32]=lds_addr, [120:64]=global_addr, [127:126]=2
//  group1: data_size=1 (2B), tensor_dim0/1, tile_dim0/1, tensor_dim0_stride
// All dims in data_size (element) units. Issued by one wave; TDM ignores EXEC.
// ---------------------------------------------------------------------------
__device__ __forceinline__ void tdm_load_2d(unsigned lds_byte_addr,
                                            const unsigned short* gsrc,
                                            unsigned tensor_d0,
                                            unsigned tensor_d1,
                                            unsigned tile_d0,
                                            unsigned tile_d1,
                                            unsigned stride0) {
  unsigned long long ga = (unsigned long long)(uintptr_t)gsrc;
  v4u g0;
  g0[0] = 1u;                                        // count=1 (valid user D#)
  g0[1] = lds_byte_addr;                             // lds_addr (bytes)
  g0[2] = (unsigned)(ga & 0xFFFFFFFFu);              // global_addr[31:0]
  g0[3] = (unsigned)((ga >> 32) & 0x01FFFFFFu)       // global_addr[56:32]
          | (2u << 30);                              // type = 2 ("image")
  v8i g1;
  g1[0] = (int)(1u << 16);                           // data_size=1 -> 2 bytes
  g1[1] = (int)((tensor_d0 & 0xFFFFu) << 16);        // tensor_dim0[15:0]
  g1[2] = (int)(((tensor_d0 >> 16) & 0xFFFFu)        // tensor_dim0[31:16]
          | ((tensor_d1 & 0xFFFFu) << 16));          // tensor_dim1[15:0]
  g1[3] = (int)(((tensor_d1 >> 16) & 0xFFFFu)        // tensor_dim1[31:16]
          | ((tile_d0 & 0xFFFFu) << 16));            // tile_dim0
  g1[4] = (int)(tile_d1 & 0xFFFFu);                  // tile_dim1; tile_dim2=0
  g1[5] = (int)stride0;                              // tensor_dim0_stride[31:0]
  g1[6] = 0;
  g1[7] = 0;
  v4i z = {0, 0, 0, 0};
#if __clang_major__ >= 23
  v8i z8 = {0, 0, 0, 0, 0, 0, 0, 0};
  __builtin_amdgcn_tensor_load_to_lds(g0, g1, z, z, z8, 0);
#else
  __builtin_amdgcn_tensor_load_to_lds(g0, g1, z, z, 0);
#endif
}

// ---------------------------------------------------------------------------
// f32 -> bf16 conversion
// ---------------------------------------------------------------------------
__global__ __launch_bounds__(256) void cvt_bf16(const float* __restrict__ in,
                                                unsigned short* __restrict__ out,
                                                int n) {
  int i = blockIdx.x * 256 + threadIdx.x;
  if (i < n) out[i] = f2bf_u(in[i]);
}

// ---------------------------------------------------------------------------
// QKV GEMM: qkv[n,f] = sum_c x[n,c] * W[f,c]  (NT, dot along c)
// Block = 8 waves = 128(M) x 64(N) tile. Per 32-wide k-step the TDM stages
// A (128x32) and W (64x32) bf16 tiles into double-buffered LDS; all 8 waves
// read fragments from LDS (W reuse x8). Q,K stored [B,H,T,D]; V stored
// transposed [B,H,D,T] so attention fragments load contiguously.
// ---------------------------------------------------------------------------
#define A_ELEMS (128 * 32)
#define W_ELEMS (64 * 32)

__global__ __launch_bounds__(256) void qkv_gemm(
    const unsigned short* __restrict__ xb,   // [4096,1024] bf16
    const unsigned short* __restrict__ wb,   // [3072,1024] bf16
    unsigned short* __restrict__ Qb,         // [B,H,T,D]
    unsigned short* __restrict__ Kb,         // [B,H,T,D]
    unsigned short* __restrict__ Vtb)        // [B,H,D,T]
{
  __shared__ unsigned short smem[2 * A_ELEMS + 2 * W_ELEMS];  // 24 KB

  const int lane = threadIdx.x & 31;
  const int wave = threadIdx.x >> 5;
  const int lh   = lane & 15;
  const int sel  = lane >> 4;               // 0: lanes 0-15, 1: lanes 16-31
  const int m0b  = blockIdx.x * 128;        // block row base (t across B*T)
  const int m0   = m0b + wave * 16;         // this wave's row tile
  const int n0   = blockIdx.y * 64;         // col tile (f)

  // TDM writes LDS behind the compiler's back. Two measures keep the LDS
  // fragment reads alive as ds_loads (addrspace(3) provenance preserved):
  //  (1) ptrtoint of smem (captures the global; the side-effecting TDM
  //      intrinsic may then write through it). Low 32 bits of a flat LDS
  //      pointer are the workgroup LDS byte offset -> descriptor lds_addr.
  //  (2) a dynamically-dead store the compiler cannot prove unreachable,
  //      so "load of never-stored LDS global == poison" cannot fire.
  const unsigned ldsb = (unsigned)(uintptr_t)smem;
  if (m0b > 0x40000000) smem[threadIdx.x] = (unsigned short)lane;

  unsigned short* bufA = smem;                    // [2][128][32]
  unsigned short* bufW = smem + 2 * A_ELEMS;      // [2][64][32]
  const unsigned offA0 = ldsb;
  const unsigned offW0 = ldsb + (unsigned)(2 * A_ELEMS * sizeof(unsigned short));

  v8f acc[4];
  for (int s = 0; s < 4; ++s)
    for (int r = 0; r < 8; ++r) acc[s][r] = 0.f;

  // Prefetch k-step 0 (wave 0 issues; TENSORcnt is per-wave).
  if (wave == 0) {
    tdm_load_2d(offA0, xb + (size_t)m0b * CDIM, CDIM, NROWS, 32, 128, CDIM);
    tdm_load_2d(offW0, wb + (size_t)n0 * CDIM, CDIM, FDIM, 32, 64, CDIM);
  }
  asm volatile("" ::: "memory");            // TDM issue not movable

  for (int i = 0; i < CDIM / 32; ++i) {
    const int cur = i & 1, nxt = cur ^ 1;
    if (wave == 0) __builtin_amdgcn_s_wait_tensorcnt(0);
    __syncthreads();                        // LDS buf[cur] visible to all
    // Order the fragment loads after the wait+barrier (TDM wrote LDS).
    asm volatile("" ::: "memory");

    if (wave == 0 && i + 1 < CDIM / 32) {   // prefetch next k-step
      const int c1 = (i + 1) * 32;
      tdm_load_2d(offA0 + (unsigned)(nxt * A_ELEMS * 2),
                  xb + (size_t)m0b * CDIM + c1, CDIM, NROWS, 32, 128, CDIM);
      tdm_load_2d(offW0 + (unsigned)(nxt * W_ELEMS * 2),
                  wb + (size_t)n0 * CDIM + c1, CDIM, FDIM, 32, 64, CDIM);
    }
    asm volatile("" ::: "memory");

    // A fragment 16x32 (M=t,K=c) from LDS: lane<16: K 0..7,16..23 ; +8 upper
    const uint4* pa =
        (const uint4*)(bufA + cur * A_ELEMS + (wave * 16 + lh) * 32 + sel * 8);
    v16bf af = pack_frag(pa[0], pa[2]);
    #pragma unroll
    for (int s = 0; s < 4; ++s) {
      // B fragment 32x16 (K=c,N=f): lane<16: K 0..7,16..23 ; +8 upper
      const uint4* pb =
          (const uint4*)(bufW + cur * W_ELEMS + (s * 16 + lh) * 32 + sel * 8);
      v16bf bf = pack_frag(pb[0], pb[2]);
      acc[s] = __builtin_amdgcn_wmma_f32_16x16x32_bf16(
          false, af, false, bf, (short)0, acc[s], false, false);
    }
    __syncthreads();                        // reads done before buf reuse
  }

  // Epilogue: D-layout: lane holds column f = n0+s*16+lh, rows t = m0+r(+8).
  const int which = n0 >> 10;          // 0=Q, 1=K, 2=V
  const int h     = (n0 & 1023) >> 6;
  const int b     = m0 >> 11;
  const int t0    = m0 & (SEQ - 1);

  if (which == 2) {
    // V transposed: [B,H,D,T]; each lane owns 8 consecutive t -> b128 store
    unsigned short* base = Vtb + (size_t)(b * HEADS + h) * HDIM * SEQ;
    for (int s = 0; s < 4; ++s) {
      const int d = s * 16 + lh;
      unsigned short* dst = base + (size_t)d * SEQ + t0 + sel * 8;
      uint4 w;
      w.x = (unsigned)f2bf_u(acc[s][0]) | ((unsigned)f2bf_u(acc[s][1]) << 16);
      w.y = (unsigned)f2bf_u(acc[s][2]) | ((unsigned)f2bf_u(acc[s][3]) << 16);
      w.z = (unsigned)f2bf_u(acc[s][4]) | ((unsigned)f2bf_u(acc[s][5]) << 16);
      w.w = (unsigned)f2bf_u(acc[s][6]) | ((unsigned)f2bf_u(acc[s][7]) << 16);
      *(uint4*)dst = w;
    }
  } else {
    unsigned short* base =
        (which == 0 ? Qb : Kb) + (size_t)(b * HEADS + h) * SEQ * HDIM;
    for (int s = 0; s < 4; ++s) {
      const int d = s * 16 + lh;
      for (int r = 0; r < 8; ++r) {
        const int t = t0 + sel * 8 + r;
        base[(size_t)t * HDIM + d] = f2bf_u(acc[s][r]);
      }
    }
  }
}

// ---------------------------------------------------------------------------
// Flash attention. One wave per (b, h, 16-query tile). Key blocks of 32.
// Computes S^T = K * Q^T and O^T = V^T * P^T so that softmax row-stats are
// per-lane (+ one xor-16 shuffle) and P^T relayout needs one half-swap.
// ---------------------------------------------------------------------------
__global__ __launch_bounds__(256) void attn_fwd(
    const unsigned short* __restrict__ Qb,   // [B,H,T,D]
    const unsigned short* __restrict__ Kb,   // [B,H,T,D]
    const unsigned short* __restrict__ Vtb,  // [B,H,D,T]
    float* __restrict__ y)                   // [B,T,C]
{
  const int lane = threadIdx.x & 31;
  const int wave = threadIdx.x >> 5;
  const int wid  = blockIdx.x * 8 + wave;    // 0 .. B*H*(T/16)-1
  const int qt   = wid & 127;                // T/16 = 128
  const int h    = (wid >> 7) & (HEADS - 1);
  const int b    = wid >> 11;
  const int q0   = qt << 4;
  const int lh   = lane & 15;
  const int sel  = lane >> 4;
  const int q    = q0 + lh;                  // this lane's query column

  const unsigned short* Qh = Qb  + (size_t)(b * HEADS + h) * SEQ * HDIM;
  const unsigned short* Kh = Kb  + (size_t)(b * HEADS + h) * SEQ * HDIM;
  const unsigned short* Vh = Vtb + (size_t)(b * HEADS + h) * HDIM * SEQ;

  // Q^T B-fragments (K-dim = d): frag j covers d = j*32..j*32+31.
  v16bf qf[2];
  #pragma unroll
  for (int j = 0; j < 2; ++j) {
    const uint4* p = (const uint4*)(Qh + (size_t)q * HDIM + j * 32 + sel * 8);
    qf[j] = pack_frag(p[0], p[2]);
  }

  // O^T accumulators: 4 d-subtiles of 16x16 (M=d, N=q).
  v8f o[4];
  for (int s = 0; s < 4; ++s)
    for (int r = 0; r < 8; ++r) o[s][r] = 0.f;
  float m = -1e30f, l = 0.f;

  const int nkb = (q0 + 47) >> 5;            // causal: keys <= q0+15
  for (int i = 0; i < nkb; ++i) {
    const int kb = i << 5;

    // K A-fragments: mt = key subtile, j = k-step over d.
    v16bf kf[2][2];
    #pragma unroll
    for (int mt = 0; mt < 2; ++mt) {
      const int key = kb + mt * 16 + lh;
      #pragma unroll
      for (int j = 0; j < 2; ++j) {
        const uint4* p =
            (const uint4*)(Kh + (size_t)key * HDIM + j * 32 + sel * 8);
        kf[mt][j] = pack_frag(p[0], p[2]);
      }
    }

    // S^T tiles (keys x queries), K-dim = D = 64 -> 2 WMMA each.
    v8f st[2];
    #pragma unroll
    for (int mt = 0; mt < 2; ++mt) {
      v8f a;
      for (int r = 0; r < 8; ++r) a[r] = 0.f;
      a = __builtin_amdgcn_wmma_f32_16x16x32_bf16(
          false, kf[mt][0], false, qf[0], (short)0, a, false, false);
      a = __builtin_amdgcn_wmma_f32_16x16x32_bf16(
          false, kf[mt][1], false, qf[1], (short)0, a, false, false);
      st[mt] = a;
    }

    // scale + causal mask (branchless; key row = kb + mt*16 + sel*8 + r)
    #pragma unroll
    for (int mt = 0; mt < 2; ++mt)
      for (int r = 0; r < 8; ++r) {
        float v = st[mt][r] * 0.125f;        // 1/sqrt(64)
        const int key = kb + mt * 16 + sel * 8 + r;
        st[mt][r] = (key > q) ? -1e30f : v;
      }

    // online softmax stats: per-lane over 16 regs, then half-swap combine
    float pm = st[0][0];
    for (int r = 1; r < 8; ++r) pm = fmaxf(pm, st[0][r]);
    for (int r = 0; r < 8; ++r) pm = fmaxf(pm, st[1][r]);
    pm = fmaxf(pm, __shfl_xor(pm, 16, 32));
    const float mn = fmaxf(m, pm);
    const float alpha = __expf(m - mn);
    m = mn;

    float ps = 0.f;
    #pragma unroll
    for (int mt = 0; mt < 2; ++mt)
      for (int r = 0; r < 8; ++r) {
        const float p = __expf(st[mt][r] - mn);
        st[mt][r] = p;
        ps += p;
      }
    ps += __shfl_xor(ps, 16, 32);
    l = l * alpha + ps;
    for (int s = 0; s < 4; ++s)
      for (int r = 0; r < 8; ++r) o[s][r] *= alpha;

    // P^T B-fragment (K=32 keys, N=16 queries): one half-swap per register.
    unsigned short e[16];
    #pragma unroll
    for (int r = 0; r < 8; ++r) {
      const float own0 = st[0][r], own1 = st[1][r];
      const float sw0 = __shfl_xor(own0, 16, 32);
      const float sw1 = __shfl_xor(own1, 16, 32);
      e[r]     = f2bf_u(sel ? sw1 : own0);   // keys kb+{0..7} / kb+{16..23}
      e[8 + r] = f2bf_u(sel ? own1 : sw0);   // keys kb+{8..15} / kb+{24..31}
    }
    v8u pt;
    #pragma unroll
    for (int d = 0; d < 8; ++d)
      pt[d] = (unsigned)e[2 * d] | ((unsigned)e[2 * d + 1] << 16);
    const v16bf pf = __builtin_bit_cast(v16bf, pt);

    // O^T += V^T(A) * P^T(B); V^T rows are contiguous along t.
    #pragma unroll
    for (int s = 0; s < 4; ++s) {
      const int d = s * 16 + lh;
      const uint4* p = (const uint4*)(Vh + (size_t)d * SEQ + kb + sel * 16);
      const v16bf vf = pack_frag(p[0], p[1]);
      o[s] = __builtin_amdgcn_wmma_f32_16x16x32_bf16(
          false, vf, false, pf, (short)0, o[s], false, false);
    }
  }

  // Normalize and store: lane owns column q; rows d = s*16 + sel*8 + r.
  const float inv = 1.0f / l;
  float* dstrow = y + ((size_t)b * SEQ + q) * CDIM + h * HDIM;
  #pragma unroll
  for (int s = 0; s < 4; ++s) {
    float4 w0, w1;
    w0.x = o[s][0] * inv; w0.y = o[s][1] * inv;
    w0.z = o[s][2] * inv; w0.w = o[s][3] * inv;
    w1.x = o[s][4] * inv; w1.y = o[s][5] * inv;
    w1.z = o[s][6] * inv; w1.w = o[s][7] * inv;
    float* dst = dstrow + s * 16 + sel * 8;
    *(float4*)dst = w0;
    *((float4*)dst + 1) = w1;
  }
}

// ---------------------------------------------------------------------------
extern "C" void kernel_launch(void* const* d_in, const int* in_sizes, int n_in,
                              void* d_out, int out_size, void* d_ws,
                              size_t ws_size, hipStream_t stream) {
  const float* x = (const float*)d_in[0];      // [B,T,C] f32
  const float* W = (const float*)d_in[1];      // [3C,C]  f32
  float* y = (float*)d_out;                    // [B,T,C] f32

  unsigned short* xb  = (unsigned short*)d_ws;               // 4096*1024
  unsigned short* wbf = xb  + (size_t)NROWS * CDIM;          // 3072*1024
  unsigned short* Qb  = wbf + (size_t)FDIM * CDIM;           // B,H,T,D
  unsigned short* Kb  = Qb  + (size_t)BATCH * HEADS * SEQ * HDIM;
  unsigned short* Vtb = Kb  + (size_t)BATCH * HEADS * SEQ * HDIM;

  const int nx = NROWS * CDIM;   // 4,194,304
  const int nw = FDIM * CDIM;    // 3,145,728
  cvt_bf16<<<(nx + 255) / 256, 256, 0, stream>>>(x, xb, nx);
  cvt_bf16<<<(nw + 255) / 256, 256, 0, stream>>>(W, wbf, nw);

  dim3 gg(NROWS / 128, FDIM / 64);             // (32, 48)
  qkv_gemm<<<gg, 256, 0, stream>>>(xb, wbf, Qb, Kb, Vtb);

  const int nwaves = BATCH * HEADS * (SEQ / 16);   // 4096
  attn_fwd<<<nwaves / 8, 256, 0, stream>>>(Qb, Kb, Vtb, y);
}
